// MultiheadAttention_53128745451523
// MI455X (gfx1250) — matrix operand order repaired and checked by reading, not compiled, not run on hardware
//
#include <hip/hip_runtime.h>

typedef __attribute__((ext_vector_type(16))) _Float16 v16h;
typedef __attribute__((ext_vector_type(8)))  float    v8f;

#define T_DIM 2048
#define B_DIM 2
#define E_DIM 1024
#define H_DIM 16
#define HD    64
#define NTOK  (T_DIM * B_DIM)   // 4096

#define WMMA_F16(a, b, c) \
  __builtin_amdgcn_wmma_f32_16x16x32_f16(false, (a), false, (b), (short)0, (c), false, false)

// ---- CDNA5 async global->LDS copy (ASYNCcnt-tracked, bypasses VGPRs) --------
__device__ __forceinline__ void async_ld_b128(const void* g, void* l) {
  // LDS aperture: generic pointer's low 32 bits are the LDS byte address.
  asm volatile("global_load_async_to_lds_b128 %0, %1, off"
               :: "v"((unsigned)(unsigned long long)l),
                  "v"((unsigned long long)g)
               : "memory");
}
template <int N>
__device__ __forceinline__ void wait_async() {
  asm volatile("s_wait_asynccnt %0" :: "n"(N) : "memory");
}

// ---- WMMA fragment loaders (wave32, f16 16x16x32) ---------------------------
// A 16x32 (MxK): lanes 0-15 -> M=lane, kBase 0; lanes 16-31 -> M=lane-16, kBase 16.
__device__ __forceinline__ v16h load_frag_a(const _Float16* s, int ld, int lane) {
  int m  = lane & 15;
  int kb = (lane >> 4) << 4;
  const _Float16* row = s + m * ld;
  v16h a;
#pragma unroll
  for (int v = 0; v < 8; ++v) {
    int k = kb + (v << 1);
    a[2 * v]     = row[k];
    a[2 * v + 1] = row[k + 1];
  }
  return a;
}

// B 32x16 (KxN), element (k,n) stored at s[n*ld + k] (k-contiguous per column).
__device__ __forceinline__ v16h load_frag_b(const _Float16* s, int ld, int lane) {
  int n  = lane & 15;
  int kb = (lane >> 4) << 4;
  const _Float16* col = s + n * ld + kb;
  v16h b;
#pragma unroll
  for (int v = 0; v < 8; ++v) {
    b[2 * v]     = col[2 * v];
    b[2 * v + 1] = col[2 * v + 1];
  }
  return b;
}

// B 32x16 (KxN), element (k,n) stored at s[k*ld + n] (row-major K x N tile).
__device__ __forceinline__ v16h load_frag_b_strided(const _Float16* s, int ld, int lane) {
  int n  = lane & 15;
  int kb = (lane >> 4) << 4;
  const _Float16* col = s + n;
  v16h b;
#pragma unroll
  for (int v = 0; v < 8; ++v) {
    b[2 * v]     = col[(kb + 2 * v) * ld];
    b[2 * v + 1] = col[(kb + 2 * v + 1) * ld];
  }
  return b;
}

// ---- fp32 -> f16 conversion pass -------------------------------------------
__global__ __launch_bounds__(256) void cvt_f16_kernel(const float* __restrict__ src,
                                                      _Float16* __restrict__ dst, int n) {
  int i = blockIdx.x * 256 + threadIdx.x;
  if (i < n) dst[i] = (_Float16)src[i];
}

// ---- GEMM: C[N,1024] = X f16 @ W^T f16, async double-buffered LDS tiles -----
__global__ __launch_bounds__(128) void gemm_f16_kernel(const _Float16* __restrict__ X,
                                                       const _Float16* __restrict__ W,
                                                       _Float16* __restrict__ Ch,
                                                       float* __restrict__ Cf,
                                                       float scale) {
  __shared__ alignas(16) _Float16 Xs[2][64 * 64];
  __shared__ alignas(16) _Float16 Ws[2][64 * 64];
  int tid  = threadIdx.x;
  int lane = tid & 31, wave = tid >> 5;
  int rowBase = blockIdx.y * 64;
  int colBase = blockIdx.x * 64;

  auto issue = [&](int kt, int buf) {
#pragma unroll
    for (int c = 0; c < 4; ++c) {
      int unit = c * 128 + tid;                 // 512 x 16B units per tile
      int r = unit >> 3, c8 = (unit & 7) * 8;
      async_ld_b128(X + (size_t)(rowBase + r) * E_DIM + kt * 64 + c8,
                    &Xs[buf][r * 64 + c8]);
      async_ld_b128(W + (size_t)(colBase + r) * E_DIM + kt * 64 + c8,
                    &Ws[buf][r * 64 + c8]);
    }
  };

  v8f acc[4] = {};
  const int NK = E_DIM / 64;                    // 16 k-steps
  issue(0, 0);
  for (int kt = 0; kt < NK; ++kt) {
    int buf = kt & 1;
    if (kt + 1 < NK) { issue(kt + 1, buf ^ 1); wait_async<8>(); }
    else             { wait_async<0>(); }
    __syncthreads();
    v16h a0 = load_frag_a(&Xs[buf][wave * 16 * 64], 64, lane);
    v16h a1 = load_frag_a(&Xs[buf][wave * 16 * 64 + 32], 64, lane);
#pragma unroll
    for (int nt = 0; nt < 4; ++nt) {
      v16h b0 = load_frag_b(&Ws[buf][nt * 16 * 64], 64, lane);
      v16h b1 = load_frag_b(&Ws[buf][nt * 16 * 64 + 32], 64, lane);
      acc[nt] = WMMA_F16(a0, b0, acc[nt]);
      acc[nt] = WMMA_F16(a1, b1, acc[nt]);
    }
    __syncthreads();
  }

  int mo = (lane >> 4) << 3;
  int n  = lane & 15;
#pragma unroll
  for (int nt = 0; nt < 4; ++nt)
#pragma unroll
    for (int r = 0; r < 8; ++r) {
      size_t idx = (size_t)(rowBase + wave * 16 + mo + r) * E_DIM + colBase + nt * 16 + n;
      float v = acc[nt][r] * scale;
      if (Cf) Cf[idx] = v;
      else    Ch[idx] = (_Float16)v;
    }
}

// ---- RoPE in place on f16 q,k (fp32 math) ----------------------------------
__global__ __launch_bounds__(256) void rope_kernel(_Float16* __restrict__ Q,
                                                   _Float16* __restrict__ K) {
  int idx = blockIdx.x * blockDim.x + threadIdx.x;
  if (idx >= NTOK * H_DIM * 32) return;
  int j = idx & 31;
  int h = (idx >> 5) & 15;
  int n = idx >> 9;                    // token row = t*B + b
  int t = n / B_DIM;
  float inv_freq = __powf(10000.0f, -(float)j * (1.0f / 32.0f));
  float ang = (float)t * inv_freq;
  float c, s;
  __sincosf(ang, &c, &s);
  size_t base = (size_t)n * E_DIM + h * HD + j;
  float q1 = (float)Q[base], q2 = (float)Q[base + 32];
  Q[base]      = (_Float16)(q1 * c - q2 * s);
  Q[base + 32] = (_Float16)(q2 * c + q1 * s);
  float k1 = (float)K[base], k2 = (float)K[base + 32];
  K[base]      = (_Float16)(k1 * c - k2 * s);
  K[base + 32] = (_Float16)(k2 * c + k1 * s);
}

// ---- Flash attention: per (b,h), 64 queries/block, async double-buffered KV -
__global__ __launch_bounds__(128) void attn_kernel(const _Float16* __restrict__ Q,
                                                   const _Float16* __restrict__ K,
                                                   const _Float16* __restrict__ V,
                                                   const float* __restrict__ mask,
                                                   _Float16* __restrict__ ctx) {
  __shared__ alignas(16) _Float16 Qs[64 * 64];       // [qrow][d]
  __shared__ alignas(16) _Float16 Ks[2][32 * 64];    // [j][d]
  __shared__ alignas(16) _Float16 Vs[2][32 * 64];    // [j][d]
  __shared__ alignas(16) _Float16 Ps[4][16 * 32];    // per wave [qrow][j]
  int tid  = threadIdx.x;
  int lane = tid & 31, wave = tid >> 5;
  int qBase = blockIdx.x * 64;
  int bh = blockIdx.y;
  int b = bh >> 4, h = bh & 15;
  size_t headOff = (size_t)h * HD;

  auto issueKV = [&](int j0, int buf) {
#pragma unroll
    for (int c = 0; c < 2; ++c) {
      int unit = c * 128 + tid;                 // 256 x 16B units per 32x64 tile
      int r = unit >> 3, c8 = (unit & 7) * 8;
      size_t g = ((size_t)(j0 + r) * B_DIM + b) * E_DIM + headOff + c8;
      async_ld_b128(K + g, &Ks[buf][r * 64 + c8]);
      async_ld_b128(V + g, &Vs[buf][r * 64 + c8]);
    }
  };

#pragma unroll
  for (int c = 0; c < 4; ++c) {                 // Q tile: 512 x 16B units
    int unit = c * 128 + tid;
    int r = unit >> 3, c8 = (unit & 7) * 8;
    async_ld_b128(Q + ((size_t)(qBase + r) * B_DIM + b) * E_DIM + headOff + c8,
                  &Qs[r * 64 + c8]);
  }
  issueKV(0, 0);
  wait_async<4>();                              // this thread's 4 Q copies done
  __syncthreads();
  v16h aq0 = load_frag_a(Qs + wave * 16 * 64, 64, lane);
  v16h aq1 = load_frag_a(Qs + wave * 16 * 64 + 32, 64, lane);

  v8f accO[4] = {};
  float mrow[8], lrow[8];
#pragma unroll
  for (int r = 0; r < 8; ++r) { mrow[r] = -1e30f; lrow[r] = 0.0f; }

  int mo = (lane >> 4) << 3;
  int n  = lane & 15;
  const float* maskBase = mask + (size_t)bh * T_DIM * T_DIM;

  for (int j0 = 0; j0 < T_DIM; j0 += 32) {
    int buf = (j0 >> 5) & 1;
    if (j0 + 32 < T_DIM) { issueKV(j0 + 32, buf ^ 1); wait_async<4>(); }
    else                 { wait_async<0>(); }
    __syncthreads();

    v8f s0 = {}, s1 = {};
    s0 = WMMA_F16(aq0, load_frag_b(&Ks[buf][0], 64, lane), s0);
    s0 = WMMA_F16(aq1, load_frag_b(&Ks[buf][32], 64, lane), s0);
    s1 = WMMA_F16(aq0, load_frag_b(&Ks[buf][16 * 64], 64, lane), s1);
    s1 = WMMA_F16(aq1, load_frag_b(&Ks[buf][16 * 64 + 32], 64, lane), s1);

    float p0[8], p1[8], alpha[8];
#pragma unroll
    for (int r = 0; r < 8; ++r) {
      int qi = qBase + wave * 16 + mo + r;
      const float* mrowp = maskBase + (size_t)qi * T_DIM;
      float v0 = s0[r] + mrowp[j0 + n];
      float v1 = s1[r] + mrowp[j0 + 16 + n];
      __builtin_prefetch(mrowp + j0 + 32 + n, 0, 1);   // next key tile (128B line)
      float mx = fmaxf(v0, v1);
#pragma unroll
      for (int off = 1; off < 16; off <<= 1)
        mx = fmaxf(mx, __shfl_xor(mx, off, 32));
      float nm = fmaxf(mrow[r], mx);
      alpha[r] = __expf(mrow[r] - nm);
      mrow[r] = nm;
      p0[r] = __expf(v0 - nm);
      p1[r] = __expf(v1 - nm);
      float rs = p0[r] + p1[r];
#pragma unroll
      for (int off = 1; off < 16; off <<= 1)
        rs += __shfl_xor(rs, off, 32);
      lrow[r] = lrow[r] * alpha[r] + rs;
    }

    _Float16* Pw = Ps[wave];
#pragma unroll
    for (int r = 0; r < 8; ++r) {
      Pw[(mo + r) * 32 + n]      = (_Float16)p0[r];
      Pw[(mo + r) * 32 + 16 + n] = (_Float16)p1[r];
#pragma unroll
      for (int nt = 0; nt < 4; ++nt) accO[nt][r] *= alpha[r];
    }
    __syncthreads();                            // C-layout P store -> A-layout load

    v16h ap = load_frag_a(Pw, 32, lane);
#pragma unroll
    for (int nt = 0; nt < 4; ++nt) {
      v16h bv = load_frag_b_strided(&Vs[buf][nt * 16], 64, lane);
      accO[nt] = WMMA_F16(ap, bv, accO[nt]);
    }
    __syncthreads();                            // readers done before buf overwrite
  }

#pragma unroll
  for (int r = 0; r < 8; ++r) {
    float inv = 1.0f / lrow[r];
    int qi = qBase + wave * 16 + mo + r;
#pragma unroll
    for (int nt = 0; nt < 4; ++nt) {
      int d = nt * 16 + n;
      ctx[((size_t)qi * B_DIM + b) * E_DIM + headOff + d] = (_Float16)(accO[nt][r] * inv);
    }
  }
}

extern "C" void kernel_launch(void* const* d_in, const int* in_sizes, int n_in,
                              void* d_out, int out_size, void* d_ws, size_t ws_size,
                              hipStream_t stream) {
  (void)in_sizes; (void)n_in; (void)out_size; (void)ws_size;
  const float* query = (const float*)d_in[0];
  const float* mask  = (const float*)d_in[1];
  const float* wq    = (const float*)d_in[2];
  const float* wk    = (const float*)d_in[3];
  const float* wv    = (const float*)d_in[4];
  const float* wo    = (const float*)d_in[5];
  float* out = (float*)d_out;

  size_t mat  = (size_t)NTOK * E_DIM;           // 4M elements
  size_t wmat = (size_t)E_DIM * E_DIM;          // 1M elements
  _Float16* x16   = (_Float16*)d_ws;
  _Float16* wq16  = x16 + mat;
  _Float16* wk16  = wq16 + wmat;
  _Float16* wv16  = wk16 + wmat;
  _Float16* wo16  = wv16 + wmat;
  _Float16* q16   = wo16 + wmat;
  _Float16* k16   = q16 + mat;
  _Float16* v16   = k16 + mat;
  _Float16* ctx16 = v16 + mat;                  // total 24M halves = 48 MB

  cvt_f16_kernel<<<(int)(mat  / 256), 256, 0, stream>>>(query, x16, (int)mat);
  cvt_f16_kernel<<<(int)(wmat / 256), 256, 0, stream>>>(wq, wq16, (int)wmat);
  cvt_f16_kernel<<<(int)(wmat / 256), 256, 0, stream>>>(wk, wk16, (int)wmat);
  cvt_f16_kernel<<<(int)(wmat / 256), 256, 0, stream>>>(wv, wv16, (int)wmat);
  cvt_f16_kernel<<<(int)(wmat / 256), 256, 0, stream>>>(wo, wo16, (int)wmat);

  dim3 gGrid(E_DIM / 64, NTOK / 64);            // (16, 64)
  gemm_f16_kernel<<<gGrid, 128, 0, stream>>>(x16, wq16, q16, nullptr, 0.125f); // hd^-0.5
  gemm_f16_kernel<<<gGrid, 128, 0, stream>>>(x16, wk16, k16, nullptr, 1.0f);
  gemm_f16_kernel<<<gGrid, 128, 0, stream>>>(x16, wv16, v16, nullptr, 1.0f);

  int rthreads = NTOK * H_DIM * 32;
  rope_kernel<<<(rthreads + 255) / 256, 256, 0, stream>>>(q16, k16);

  attn_kernel<<<dim3(T_DIM / 64, B_DIM * H_DIM), 128, 0, stream>>>(q16, k16, v16, mask, ctx16);

  gemm_f16_kernel<<<gGrid, 128, 0, stream>>>(ctx16, wo16, nullptr, out, 1.0f);
}
